// AttentionTier_60808146977376
// MI455X (gfx1250) — compile-verified, matrix operand order checked by reading, not run on hardware
//
#include <hip/hip_runtime.h>

typedef __attribute__((ext_vector_type(16))) __bf16 v16bf;
typedef __attribute__((ext_vector_type(8)))  __bf16 v8bf;
typedef __attribute__((ext_vector_type(4)))  __bf16 v4bf;
typedef __attribute__((ext_vector_type(8)))  float  v8f;
typedef __attribute__((ext_vector_type(4)))  float  v4f;

#define N_HEADS  16
#define HEAD_DIM 64
#define D_MODEL  1024
#define B_SZ     4
#define T_SEQ    2048

static __device__ __forceinline__ v16bf cat8(v8bf lo, v8bf hi) {
  v16bf r;
#pragma unroll
  for (int i = 0; i < 8; ++i) { r[i] = lo[i]; r[i + 8] = hi[i]; }
  return r;
}
static __device__ __forceinline__ v8bf lds8(const __bf16* p) {
  return *reinterpret_cast<const v8bf*>(p);
}
static __device__ __forceinline__ v8bf glb8(const __bf16* p) {
  return *reinterpret_cast<const v8bf*>(p);
}

// Issue one per-lane async 16B copy global->LDS (ASYNCcnt path, bypasses VGPRs)
static __device__ __forceinline__ void async_cp16(const __bf16* gsrc, __bf16* ldst) {
  unsigned loff = (unsigned)(unsigned long long)(void*)ldst;   // low 32b = LDS offset
  unsigned long long gaddr = (unsigned long long)(const void*)gsrc;
  asm volatile("global_load_async_to_lds_b128 %0, %1, off"
               :: "v"(loff), "v"(gaddr) : "memory");
}
static __device__ __forceinline__ void wait_async0() {
  asm volatile("s_wait_asynccnt 0x0" ::: "memory");
}

// Stage a 64(row) x 32(k) bf16 tile into LDS via async DMA (128 threads).
static __device__ __forceinline__ void stage_tile_async(
    const __bf16* gbase /* + row*ldk + c */, int ldk, __bf16* lbuf, int tid) {
#pragma unroll
  for (int i = 0; i < 2; ++i) {
    int e = tid + i * 128;          // 256 16B-chunks
    int row = e >> 2;
    int c   = (e & 3) * 8;
    async_cp16(gbase + (size_t)row * ldk + c, &lbuf[row * 32 + c]);
  }
}

// ---------------------------------------------------------------------------
// fp32 -> bf16 bulk convert (one-time, bandwidth-trivial at 23.3 TB/s)
// ---------------------------------------------------------------------------
__global__ __launch_bounds__(256)
void cvt_bf16_kernel(const float* __restrict__ src, __bf16* __restrict__ dst, int n8) {
  int i = blockIdx.x * 256 + threadIdx.x;
  if (i < n8) {
    v4f a = *reinterpret_cast<const v4f*>(src + (size_t)i * 8);
    v4f b = *reinterpret_cast<const v4f*>(src + (size_t)i * 8 + 4);
    v8bf o;
#pragma unroll
    for (int j = 0; j < 4; ++j) { o[j] = (__bf16)a[j]; o[4 + j] = (__bf16)b[j]; }
    *reinterpret_cast<v8bf*>(dst + (size_t)i * 8) = o;
  }
}

// ---------------------------------------------------------------------------
// Kernel 1: qkv = x @ W_qkv^T (bf16 WMMA, f32 accum), async double-buffered
// LDS staging, fused RoPE epilogue, scatter to Q/K/V [B,H,T,Dh] bf16.
// Block: 128 threads (4 waves), tile 64x64, K-step 32.
// ---------------------------------------------------------------------------
__global__ __launch_bounds__(128)
void qkv_rope_kernel(const __bf16* __restrict__ X, const __bf16* __restrict__ W,
                     __bf16* __restrict__ Qo, __bf16* __restrict__ Ko,
                     __bf16* __restrict__ Vo)
{
  __shared__ __bf16 Xs[2][64 * 32];
  __shared__ __bf16 Ws[2][64 * 32];
  const int tid  = threadIdx.x;
  const int lane = tid & 31;
  const int wave = tid >> 5;
  const int n16  = lane & 15;
  const int half = lane >> 4;
  const int kb8  = half * 8;
  const int m0   = blockIdx.x * 64;
  const int n0   = blockIdx.y * 64;

  const __bf16* Ab = X + (size_t)m0 * D_MODEL;
  const __bf16* Bb = W + (size_t)n0 * D_MODEL;

  v8f acc[4] = {};
  const int nsteps = D_MODEL / 32;

  stage_tile_async(Ab, D_MODEL, Xs[0], tid);
  stage_tile_async(Bb, D_MODEL, Ws[0], tid);

  for (int i = 0; i < nsteps; ++i) {
    wait_async0();          // my tile-i copies landed in LDS
    __syncthreads();        // everyone's tile-i visible; tile-(i-1) reads done
    if (i + 1 < nsteps) {   // overlap: DMA tile i+1 while consuming tile i
      stage_tile_async(Ab + (i + 1) * 32, D_MODEL, Xs[(i + 1) & 1], tid);
      stage_tile_async(Bb + (i + 1) * 32, D_MODEL, Ws[(i + 1) & 1], tid);
    }
    const __bf16* Xb = Xs[i & 1];
    const __bf16* Wb = Ws[i & 1];
    const __bf16* ar = &Xb[(wave * 16 + n16) * 32];
    v16bf afrag = cat8(lds8(ar + kb8), lds8(ar + kb8 + 16));
    v16bf bfrag[4];
#pragma unroll
    for (int nt = 0; nt < 4; ++nt) {
      const __bf16* br = &Wb[(nt * 16 + n16) * 32];
      bfrag[nt] = cat8(lds8(br + kb8), lds8(br + kb8 + 16));
    }
#pragma unroll
    for (int nt = 0; nt < 4; ++nt)
      acc[nt] = __builtin_amdgcn_wmma_f32_16x16x32_bf16(
          false, afrag, false, bfrag[nt], (short)0, acc[nt], false, false);
  }

  const int sel = n0 / D_MODEL;               // 0=q, 1=k, 2=v
  const int h   = (n0 % D_MODEL) / HEAD_DIM;
#pragma unroll
  for (int r = 0; r < 8; ++r) {
    int m = m0 + wave * 16 + r + half * 8;    // global row (b*T + t)
    int b = m >> 11;
    int t = m & (T_SEQ - 1);
    size_t base = ((size_t)(b * N_HEADS + h) * T_SEQ + t) * HEAD_DIM;
    if (sel == 2) {
#pragma unroll
      for (int nt = 0; nt < 4; ++nt)
        Vo[base + nt * 16 + n16] = (__bf16)acc[nt][r];
    } else {
      __bf16* dst = (sel == 0) ? Qo : Ko;
      const float sc = (sel == 0) ? 0.125f : 1.0f;  // 1/sqrt(64) folded into Q
#pragma unroll
      for (int nt = 0; nt < 2; ++nt) {
        int d = nt * 16 + n16;                // d < 32; partner is d+32
        float ang = (float)t * __expf(-(float)d * (9.210340371976184f / 32.0f));
        float c = __cosf(ang), s = __sinf(ang);
        float lo = acc[nt][r], hi = acc[nt + 2][r];
        dst[base + d]      = (__bf16)((lo * c - hi * s) * sc);
        dst[base + d + 32] = (__bf16)((hi * c + lo * s) * sc);
      }
    }
  }
}

// ---------------------------------------------------------------------------
// Kernel 2: causal flash attention. One block per (b*H, 64-query tile),
// 4 waves; 32 keys/iter: 4 S-WMMAs + 4 PV-WMMAs, online softmax via
// C/D-layout shuffle reductions. Output written bf16 for the async out-proj.
// ---------------------------------------------------------------------------
__global__ __launch_bounds__(128)
void attn_kernel(const __bf16* __restrict__ Qb, const __bf16* __restrict__ Kb,
                 const __bf16* __restrict__ Vb, __bf16* __restrict__ O)
{
  __shared__ __bf16 Vt[64 * 32];      // transposed V tile: [d][key]
  __shared__ __bf16 Pb[4][16 * 32];   // per-wave P re-layout buffer [m][kk]
  const int tid  = threadIdx.x;
  const int lane = tid & 31;
  const int wave = tid >> 5;
  const int n16  = lane & 15;
  const int half = lane >> 4;
  const int kb8  = half * 8;
  const int q0   = blockIdx.x * 64;
  const int bh   = blockIdx.y;
  const size_t hb = (size_t)bh * T_SEQ * HEAD_DIM;

  const __bf16* qp = Qb + hb + (size_t)(q0 + wave * 16 + n16) * HEAD_DIM;
  v16bf qa0 = cat8(glb8(qp + kb8),      glb8(qp + kb8 + 16));
  v16bf qa1 = cat8(glb8(qp + 32 + kb8), glb8(qp + 32 + kb8 + 16));

  v8f o[4] = {};
  float rm[8], rs[8];
#pragma unroll
  for (int r = 0; r < 8; ++r) { rm[r] = -1e30f; rs[r] = 0.0f; }

  const int niter = (q0 + 63) / 32 + 1;
  for (int it = 0; it < niter; ++it) {
    const int k0 = it * 32;
    __syncthreads();
#pragma unroll
    for (int i = 0; i < 4; ++i) {       // stage V[k0..k0+31][:] transposed
      int e   = tid + i * 128;
      int key = e >> 4;
      int dq  = (e & 15) * 4;
      v4bf vv = *reinterpret_cast<const v4bf*>(
          Vb + hb + (size_t)(k0 + key) * HEAD_DIM + dq);
#pragma unroll
      for (int j = 0; j < 4; ++j) Vt[(dq + j) * 32 + key] = vv[j];
    }
    __syncthreads();

    v8f S[2] = {};
#pragma unroll
    for (int kt = 0; kt < 2; ++kt) {
      const __bf16* kp = Kb + hb + (size_t)(k0 + kt * 16 + n16) * HEAD_DIM;
      v16bf b0 = cat8(glb8(kp + kb8),      glb8(kp + kb8 + 16));
      v16bf b1 = cat8(glb8(kp + 32 + kb8), glb8(kp + 32 + kb8 + 16));
      S[kt] = __builtin_amdgcn_wmma_f32_16x16x32_bf16(
          false, qa0, false, b0, (short)0, S[kt], false, false);
      S[kt] = __builtin_amdgcn_wmma_f32_16x16x32_bf16(
          false, qa1, false, b1, (short)0, S[kt], false, false);
    }

    const int qrbase = q0 + wave * 16 + half * 8;
#pragma unroll
    for (int r = 0; r < 8; ++r) {
      int qi = qrbase + r;
      float s0 = S[0][r]; if (k0 + n16 > qi)      s0 = -1e30f;
      float s1 = S[1][r]; if (k0 + 16 + n16 > qi) s1 = -1e30f;
      float mx = fmaxf(s0, s1);
#pragma unroll
      for (int off = 1; off < 16; off <<= 1) mx = fmaxf(mx, __shfl_xor(mx, off, 32));
      float nm   = fmaxf(rm[r], mx);
      float corr = __expf(rm[r] - nm);
      rm[r] = nm;
      float p0 = __expf(s0 - nm);
      float p1 = __expf(s1 - nm);
      float ps = p0 + p1;
#pragma unroll
      for (int off = 1; off < 16; off <<= 1) ps += __shfl_xor(ps, off, 32);
      rs[r] = rs[r] * corr + ps;
#pragma unroll
      for (int nt = 0; nt < 4; ++nt) o[nt][r] *= corr;
      __bf16* pr = &Pb[wave][(r + half * 8) * 32];
      pr[n16]      = (__bf16)p0;
      pr[16 + n16] = (__bf16)p1;
    }
    __syncthreads();

    const __bf16* pav = &Pb[wave][n16 * 32];
    v16bf pa = cat8(lds8(pav + kb8), lds8(pav + kb8 + 16));
    v16bf vf[4];
#pragma unroll
    for (int nt = 0; nt < 4; ++nt) {
      const __bf16* vr = &Vt[(nt * 16 + n16) * 32];
      vf[nt] = cat8(lds8(vr + kb8), lds8(vr + kb8 + 16));
    }
#pragma unroll
    for (int nt = 0; nt < 4; ++nt)
      o[nt] = __builtin_amdgcn_wmma_f32_16x16x32_bf16(
          false, pa, false, vf[nt], (short)0, o[nt], false, false);
  }

  const int b = bh / N_HEADS, h = bh % N_HEADS;
#pragma unroll
  for (int r = 0; r < 8; ++r) {
    float inv = 1.0f / rs[r];
    int t = q0 + wave * 16 + half * 8 + r;
#pragma unroll
    for (int nt = 0; nt < 4; ++nt) {
      int d = nt * 16 + n16;
      O[(size_t)(b * T_SEQ + t) * D_MODEL + h * HEAD_DIM + d] =
          (__bf16)(o[nt][r] * inv);
    }
  }
}

// ---------------------------------------------------------------------------
// Kernel 3: out = O @ W_out^T  (bf16 async-staged WMMA GEMM, f32 output)
// ---------------------------------------------------------------------------
__global__ __launch_bounds__(128)
void out_proj_kernel(const __bf16* __restrict__ A, const __bf16* __restrict__ W,
                     float* __restrict__ Out)
{
  __shared__ __bf16 Xs[2][64 * 32];
  __shared__ __bf16 Ws[2][64 * 32];
  const int tid  = threadIdx.x;
  const int lane = tid & 31;
  const int wave = tid >> 5;
  const int n16  = lane & 15;
  const int half = lane >> 4;
  const int kb8  = half * 8;
  const int m0   = blockIdx.x * 64;
  const int n0   = blockIdx.y * 64;

  const __bf16* Ab = A + (size_t)m0 * D_MODEL;
  const __bf16* Bb = W + (size_t)n0 * D_MODEL;

  v8f acc[4] = {};
  const int nsteps = D_MODEL / 32;

  stage_tile_async(Ab, D_MODEL, Xs[0], tid);
  stage_tile_async(Bb, D_MODEL, Ws[0], tid);

  for (int i = 0; i < nsteps; ++i) {
    wait_async0();
    __syncthreads();
    if (i + 1 < nsteps) {
      stage_tile_async(Ab + (i + 1) * 32, D_MODEL, Xs[(i + 1) & 1], tid);
      stage_tile_async(Bb + (i + 1) * 32, D_MODEL, Ws[(i + 1) & 1], tid);
    }
    const __bf16* Xb = Xs[i & 1];
    const __bf16* Wb = Ws[i & 1];
    const __bf16* ar = &Xb[(wave * 16 + n16) * 32];
    v16bf afrag = cat8(lds8(ar + kb8), lds8(ar + kb8 + 16));
    v16bf bfrag[4];
#pragma unroll
    for (int nt = 0; nt < 4; ++nt) {
      const __bf16* br = &Wb[(nt * 16 + n16) * 32];
      bfrag[nt] = cat8(lds8(br + kb8), lds8(br + kb8 + 16));
    }
#pragma unroll
    for (int nt = 0; nt < 4; ++nt)
      acc[nt] = __builtin_amdgcn_wmma_f32_16x16x32_bf16(
          false, afrag, false, bfrag[nt], (short)0, acc[nt], false, false);
  }

#pragma unroll
  for (int r = 0; r < 8; ++r) {
    int m = m0 + wave * 16 + r + half * 8;
#pragma unroll
    for (int nt = 0; nt < 4; ++nt)
      Out[(size_t)m * D_MODEL + n0 + nt * 16 + n16] = acc[nt][r];
  }
}

// ---------------------------------------------------------------------------
extern "C" void kernel_launch(void* const* d_in, const int* in_sizes, int n_in,
                              void* d_out, int out_size, void* d_ws, size_t ws_size,
                              hipStream_t stream)
{
  const float* x    = (const float*)d_in[0];   // [4,2048,1024]
  const float* Wqkv = (const float*)d_in[1];   // [3072,1024]
  const float* Wout = (const float*)d_in[2];   // [1024,1024]
  float* out = (float*)d_out;                  // [4,2048,1024]

  const size_t bhtd  = (size_t)B_SZ * N_HEADS * T_SEQ * HEAD_DIM;   // 8Mi
  const size_t n_x   = (size_t)B_SZ * T_SEQ * D_MODEL;              // 8Mi
  const size_t n_wq  = (size_t)3 * D_MODEL * D_MODEL;               // 3Mi
  const size_t n_wo  = (size_t)D_MODEL * D_MODEL;                   // 1Mi

  __bf16* Qr   = (__bf16*)d_ws;
  __bf16* Kr   = Qr + bhtd;
  __bf16* Vr   = Kr + bhtd;
  __bf16* Xbf  = Vr + bhtd;
  __bf16* Wqb  = Xbf + n_x;
  __bf16* Wob  = Wqb + n_wq;
  __bf16* Ob   = Wob + n_wo;     // attention output, bf16 [4,2048,1024]

  dim3 blk(128);
  cvt_bf16_kernel<<<(int)(n_x  / 8 + 255) / 256, 256, 0, stream>>>(x,    Xbf, (int)(n_x  / 8));
  cvt_bf16_kernel<<<(int)(n_wq / 8 + 255) / 256, 256, 0, stream>>>(Wqkv, Wqb, (int)(n_wq / 8));
  cvt_bf16_kernel<<<(int)(n_wo / 8 + 255) / 256, 256, 0, stream>>>(Wout, Wob, (int)(n_wo / 8));

  qkv_rope_kernel<<<dim3(128, 48), blk, 0, stream>>>(Xbf, Wqb, Qr, Kr, Vr);
  attn_kernel    <<<dim3(32, 64),  blk, 0, stream>>>(Qr, Kr, Vr, Ob);
  out_proj_kernel<<<dim3(128, 16), blk, 0, stream>>>(Ob, Wob, out);
}
